// MLP_embedding_layer_4312147165641
// MI455X (gfx1250) — compile-verified, matrix operand order checked by reading, not run on hardware
//
#include <hip/hip_runtime.h>

typedef __attribute__((ext_vector_type(16))) _Float16 v16h;
typedef __attribute__((ext_vector_type(8)))  _Float16 v8h;
typedef __attribute__((ext_vector_type(8)))  float    v8f;

#define B_ROWS   262144
#define NFEA     256
#define HALF_FEA 128
#define H1 64
#define H2 32
#define H3 16
#define CO 32
#define EMB 8
#define ODIM 16

#define WAVES_PER_BLOCK 4
#define THREADS (WAVES_PER_BLOCK * 32)
#define ROWS_PER_WAVE 16
#define ROWS_PER_BLOCK (WAVES_PER_BLOCK * ROWS_PER_WAVE)

// Weight fragments pre-swizzled into the CDNA5 WMMA B layout:
// frag = kc*(N/16)+t ; per lane 16 f16 stored contiguously (32 B).
struct Smem {
  alignas(32) _Float16 W1f[16 * 32 * 16];  // 16 KB : 128x64, 4 kc x 4 t
  alignas(32) _Float16 W2f[4 * 32 * 16];   //  4 KB : 64x32,  2 kc x 2 t
  alignas(32) _Float16 W3f[1 * 32 * 16];   //  1 KB : 32x16
  alignas(32) _Float16 Wof[1 * 32 * 16];   //  1 KB : 32x16
  float Wemb[CO * EMB];
  float Bemb[CO * EMB];
  float Mm[EMB * EMB];
  float WembM[16 * EMB];  // Wemb[0:16] @ M   (left-side emb folded with M)
  float BembM[16 * EMB];  // Bemb[0:16] @ M
  float b1v[H1];
  float b2v[H2];
  float b3v[H3];
  float bov[ODIM];
  alignas(16) _Float16 h0[WAVES_PER_BLOCK][ROWS_PER_WAVE * H1];  // 8 KB
  alignas(16) _Float16 h1[WAVES_PER_BLOCK][ROWS_PER_WAVE * H2];  // 4 KB
  alignas(16) float out32[WAVES_PER_BLOCK][ROWS_PER_WAVE * CO];  // 8 KB
};

__device__ __forceinline__ v8f wmma32f16(v16h a, v16h b, v8f c) {
  return __builtin_amdgcn_wmma_f32_16x16x32_f16(false, a, false, b, (short)0, c,
                                                false, false);
}

// Pre-swizzled fragment load: two contiguous ds_load_b128 per fragment.
__device__ __forceinline__ v16h loadFrag(const _Float16* F, int frag, int lane) {
  return *(const v16h*)(F + ((frag * 32 + lane) << 4));
}

// A fragment from row-major f16 activation [16][ldk]: two ds_load_b128.
__device__ __forceinline__ v16h gatherA_lds(const _Float16* H, int ldk, int k0,
                                            int lane) {
  int m  = lane & 15;
  int kb = k0 + ((lane >> 4) << 3);
  v8h lo = *(const v8h*)(H + m * ldk + kb);
  v8h hi = *(const v8h*)(H + m * ldk + kb + 16);
  return __builtin_shufflevector(lo, hi, 0, 1, 2, 3, 4, 5, 6, 7, 8, 9, 10, 11,
                                 12, 13, 14, 15);
}

// Swizzle a row-major f32 weight [K][N] into WMMA-B fragment layout (f16).
__device__ __forceinline__ void stageFrags(_Float16* dst, const float* src,
                                           int K, int N, int tid) {
  int nt = N >> 4;
  int total = (K >> 5) * nt * 32;  // fragments * lanes
  for (int idx = tid; idx < total; idx += THREADS) {
    int lane = idx & 31;
    int f    = idx >> 5;
    int kc   = f / nt;
    int t    = f - kc * nt;
    int n    = (t << 4) + (lane & 15);
    int kb   = (kc << 5) + ((lane >> 4) << 3);
    v16h v;
#pragma unroll
    for (int p = 0; p < 8; ++p) v[p]     = (_Float16)src[(kb + p) * N + n];
#pragma unroll
    for (int p = 0; p < 8; ++p) v[8 + p] = (_Float16)src[(kb + 16 + p) * N + n];
    *(v16h*)(dst + (idx << 4)) = v;
  }
}

// Wave-private LDS RAW ordering: LDS ops from one wave execute in order in the
// LDS pipe; we only need to stop the compiler from reordering around it.
__device__ __forceinline__ void waveFence() {
  __builtin_amdgcn_wave_barrier();
  __builtin_amdgcn_fence(__ATOMIC_ACQ_REL, "wavefront");
  __builtin_amdgcn_wave_barrier();
}

__device__ __forceinline__ float tanh_fast(float x) {
  float e = __expf(2.0f * x);  // v_exp_f32
  return (e - 1.0f) / (e + 1.0f);
}

__device__ __forceinline__ v8f biasAcc(float bias) {
  v8f a;
#pragma unroll
  for (int r = 0; r < 8; ++r) a[r] = bias;
  return a;
}

__global__ __launch_bounds__(THREADS)
void fused_twin_tower_interact(const float* __restrict__ x,
                               const float* __restrict__ W1, const float* __restrict__ b1,
                               const float* __restrict__ W2, const float* __restrict__ b2,
                               const float* __restrict__ W3, const float* __restrict__ b3,
                               const float* __restrict__ Wemb, const float* __restrict__ Bemb,
                               const float* __restrict__ Mmat, const float* __restrict__ Wo,
                               const float* __restrict__ bo, float* __restrict__ out) {
  __shared__ Smem s;
  const int tid  = threadIdx.x;
  const int lane = tid & 31;
  const int wv   = tid >> 5;
  const int rowBase = blockIdx.x * ROWS_PER_BLOCK + wv * ROWS_PER_WAVE;
  const int m   = lane & 15;
  const int lhi = lane >> 4;

  // ---- stage weights: WMMA operands pre-swizzled to fragment layout -------
  stageFrags(s.W1f, W1, HALF_FEA, H1, tid);
  stageFrags(s.W2f, W2, H1, H2, tid);
  stageFrags(s.W3f, W3, H2, H3, tid);
  stageFrags(s.Wof, Wo, CO, ODIM, tid);
  for (int i = tid; i < CO * EMB; i += THREADS) { s.Wemb[i] = Wemb[i]; s.Bemb[i] = Bemb[i]; }
  for (int i = tid; i < EMB * EMB; i += THREADS) s.Mm[i]  = Mmat[i];
  for (int i = tid; i < H1; i += THREADS)        s.b1v[i] = b1[i];
  for (int i = tid; i < H2; i += THREADS)        s.b2v[i] = b2[i];
  for (int i = tid; i < H3; i += THREADS)        s.b3v[i] = b3[i];
  for (int i = tid; i < ODIM; i += THREADS)      s.bov[i] = bo[i];
  __syncthreads();

  // Fold M into the left-side embedding: l_i@M = o_i*(Wemb_i@M) + (Bemb_i@M)
  for (int idx = tid; idx < 16 * EMB; idx += THREADS) {
    int i = idx >> 3, f = idx & 7;
    float aw = 0.f, ab = 0.f;
#pragma unroll
    for (int e = 0; e < 8; ++e) {
      aw += s.Wemb[i * EMB + e] * s.Mm[e * EMB + f];
      ab += s.Bemb[i * EMB + e] * s.Mm[e * EMB + f];
    }
    s.WembM[idx] = aw;
    s.BembM[idx] = ab;
  }
  __syncthreads();

  // =========================== twin towers ================================
  for (int side = 0; side < 2; ++side) {
    // A fragments of X (16 rows x 128 k) from HBM via b128 loads, f32 -> f16
    v16h ax[4];
    {
      const float4* rp = (const float4*)(x + (size_t)(rowBase + m) * NFEA +
                                         side * HALF_FEA + 8 * lhi);
#pragma unroll
      for (int kc = 0; kc < 4; ++kc) {
        float4 a0 = rp[8 * kc + 0], a1 = rp[8 * kc + 1];
        float4 b0 = rp[8 * kc + 4], b1 = rp[8 * kc + 5];
        ax[kc][0] = (_Float16)a0.x; ax[kc][1] = (_Float16)a0.y;
        ax[kc][2] = (_Float16)a0.z; ax[kc][3] = (_Float16)a0.w;
        ax[kc][4] = (_Float16)a1.x; ax[kc][5] = (_Float16)a1.y;
        ax[kc][6] = (_Float16)a1.z; ax[kc][7] = (_Float16)a1.w;
        ax[kc][8]  = (_Float16)b0.x; ax[kc][9]  = (_Float16)b0.y;
        ax[kc][10] = (_Float16)b0.z; ax[kc][11] = (_Float16)b0.w;
        ax[kc][12] = (_Float16)b1.x; ax[kc][13] = (_Float16)b1.y;
        ax[kc][14] = (_Float16)b1.z; ax[kc][15] = (_Float16)b1.w;
      }
    }
    // Layer 1: (16x128)@(128x64) -> 4 n-tiles x 4 k-chunks; bias rides in C
#pragma unroll
    for (int t = 0; t < 4; ++t) {
      int n = 16 * t + m;
      v8f acc = biasAcc(s.b1v[n]);
#pragma unroll
      for (int kc = 0; kc < 4; ++kc)
        acc = wmma32f16(ax[kc], loadFrag(s.W1f, kc * 4 + t, lane), acc);
#pragma unroll
      for (int r = 0; r < 8; ++r) {
        float v = acc[r] > 0.f ? acc[r] : 0.f;
        s.h0[wv][(r + 8 * lhi) * H1 + n] = (_Float16)v;
      }
    }
    waveFence();
    // Layer 2: (16x64)@(64x32) -> 2 n-tiles x 2 k-chunks
    {
      v8f acc0 = biasAcc(s.b2v[m]);
      v8f acc1 = biasAcc(s.b2v[16 + m]);
#pragma unroll
      for (int kc = 0; kc < 2; ++kc) {
        v16h af = gatherA_lds(s.h0[wv], H1, 32 * kc, lane);
        acc0 = wmma32f16(af, loadFrag(s.W2f, kc * 2 + 0, lane), acc0);
        acc1 = wmma32f16(af, loadFrag(s.W2f, kc * 2 + 1, lane), acc1);
      }
#pragma unroll
      for (int r = 0; r < 8; ++r) {
        float v0 = acc0[r] > 0.f ? acc0[r] : 0.f;
        float v1 = acc1[r] > 0.f ? acc1[r] : 0.f;
        s.h1[wv][(r + 8 * lhi) * H2 + m]      = (_Float16)v0;
        s.h1[wv][(r + 8 * lhi) * H2 + 16 + m] = (_Float16)v1;
      }
    }
    waveFence();
    // Layer 3: (16x32)@(32x16) -> single WMMA; f32 result into out32 half
    {
      v16h af = gatherA_lds(s.h1[wv], H2, 0, lane);
      v8f acc = biasAcc(s.b3v[m]);
      acc = wmma32f16(af, loadFrag(s.W3f, 0, lane), acc);
#pragma unroll
      for (int r = 0; r < 8; ++r) {
        float v = acc[r] > 0.f ? acc[r] : 0.f;
        s.out32[wv][(r + 8 * lhi) * CO + side * 16 + m] = v;
      }
    }
    waveFence();
  }

  // ======================= bilinear interaction ===========================
  // lane pair (m, lhi): this lane handles row m, j-range [8*lhi, 8*lhi+8)
  float o[CO];
  {
    const float4* orow = (const float4*)&s.out32[wv][m * CO];
#pragma unroll
    for (int q = 0; q < 8; ++q) {
      float4 v = orow[q];
      o[4 * q + 0] = v.x; o[4 * q + 1] = v.y;
      o[4 * q + 2] = v.z; o[4 * q + 3] = v.w;
    }
  }

  float rbuf[8][8];
#pragma unroll
  for (int jj = 0; jj < 8; ++jj) {
    int j = 8 * lhi + jj;
    float oj = o[16 + j];
#pragma unroll
    for (int f = 0; f < 8; ++f)
      rbuf[jj][f] = oj * s.Wemb[(16 + j) * EMB + f] + s.Bemb[(16 + j) * EMB + f];
  }

  float colsum[8];
#pragma unroll
  for (int jj = 0; jj < 8; ++jj) colsum[jj] = 0.f;
  float rs[16];

#pragma unroll
  for (int i = 0; i < 16; ++i) {
    // t_i = l_i @ M, with M pre-folded:  t_i[f] = o_i*WembM[i][f] + BembM[i][f]
    float tf[8];
#pragma unroll
    for (int f = 0; f < 8; ++f)
      tf[f] = o[i] * s.WembM[i * EMB + f] + s.BembM[i * EMB + f];
    float part = 0.f;
#pragma unroll
    for (int jj = 0; jj < 8; ++jj) {
      float g = 0.f;
#pragma unroll
      for (int f = 0; f < 8; ++f) g += tf[f] * rbuf[jj][f];
      g = tanh_fast(g);
      colsum[jj] += g;
      part += g;
    }
    rs[i] = part + __shfl_xor(part, 16, 32);  // full sum over all 16 j
  }

  // lr = softmax_j of G.sum(axis=1): 16 colsums split 8/8 across the pair
  float mx = colsum[0];
#pragma unroll
  for (int jj = 1; jj < 8; ++jj) mx = fmaxf(mx, colsum[jj]);
  mx = fmaxf(mx, __shfl_xor(mx, 16, 32));
  float ex[8], ssum = 0.f;
#pragma unroll
  for (int jj = 0; jj < 8; ++jj) { ex[jj] = __expf(colsum[jj] - mx); ssum += ex[jj]; }
  ssum += __shfl_xor(ssum, 16, 32);
  float inv = 1.f / ssum;

  // rl = softmax_i of G.sum(axis=2): all 16 values resident per lane
  float mx2 = rs[0];
#pragma unroll
  for (int i = 1; i < 16; ++i) mx2 = fmaxf(mx2, rs[i]);
  float er[16], s2 = 0.f;
#pragma unroll
  for (int i = 0; i < 16; ++i) { er[i] = __expf(rs[i] - mx2); s2 += er[i]; }
  float inv2 = 1.f / s2;

  // gate features back into LDS with b128 stores
  {
    float4* wrow = (float4*)&s.out32[wv][m * CO];
    float4 g0, g1;
    g0.x = o[8 * lhi + 0] * ex[0] * inv; g0.y = o[8 * lhi + 1] * ex[1] * inv;
    g0.z = o[8 * lhi + 2] * ex[2] * inv; g0.w = o[8 * lhi + 3] * ex[3] * inv;
    g1.x = o[8 * lhi + 4] * ex[4] * inv; g1.y = o[8 * lhi + 5] * ex[5] * inv;
    g1.z = o[8 * lhi + 6] * ex[6] * inv; g1.w = o[8 * lhi + 7] * ex[7] * inv;
    wrow[2 * lhi + 0] = g0;
    wrow[2 * lhi + 1] = g1;
    if (lhi == 0) {  // rl fully known by lane-half 0
#pragma unroll
      for (int q = 0; q < 4; ++q) {
        float4 gr;
        gr.x = o[16 + 4 * q + 0] * er[4 * q + 0] * inv2;
        gr.y = o[16 + 4 * q + 1] * er[4 * q + 1] * inv2;
        gr.z = o[16 + 4 * q + 2] * er[4 * q + 2] * inv2;
        gr.w = o[16 + 4 * q + 3] * er[4 * q + 3] * inv2;
        wrow[4 + q] = gr;
      }
    }
  }
  waveFence();

  // ====================== final 32->16 GEMM + ReLU ========================
  {
    const float4* op = (const float4*)&s.out32[wv][m * CO + 8 * lhi];
    float4 a0 = op[0], a1 = op[1], b0 = op[4], b1 = op[5];
    v16h af;
    af[0] = (_Float16)a0.x; af[1] = (_Float16)a0.y;
    af[2] = (_Float16)a0.z; af[3] = (_Float16)a0.w;
    af[4] = (_Float16)a1.x; af[5] = (_Float16)a1.y;
    af[6] = (_Float16)a1.z; af[7] = (_Float16)a1.w;
    af[8]  = (_Float16)b0.x; af[9]  = (_Float16)b0.y;
    af[10] = (_Float16)b0.z; af[11] = (_Float16)b0.w;
    af[12] = (_Float16)b1.x; af[13] = (_Float16)b1.y;
    af[14] = (_Float16)b1.z; af[15] = (_Float16)b1.w;
    v8f acc = biasAcc(s.bov[m]);
    acc = wmma32f16(af, loadFrag(s.Wof, 0, lane), acc);
#pragma unroll
    for (int r = 0; r < 8; ++r) {
      float v = acc[r] > 0.f ? acc[r] : 0.f;
      out[(size_t)(rowBase + r + 8 * lhi) * ODIM + m] = v;
    }
  }
}

extern "C" void kernel_launch(void* const* d_in, const int* in_sizes, int n_in,
                              void* d_out, int out_size, void* d_ws, size_t ws_size,
                              hipStream_t stream) {
  const float* x    = (const float*)d_in[0];
  const float* W1   = (const float*)d_in[1];
  const float* b1   = (const float*)d_in[2];
  const float* W2   = (const float*)d_in[3];
  const float* b2   = (const float*)d_in[4];
  const float* W3   = (const float*)d_in[5];
  const float* b3   = (const float*)d_in[6];
  const float* Wemb = (const float*)d_in[7];
  const float* Bemb = (const float*)d_in[8];
  const float* Mm   = (const float*)d_in[9];
  const float* Wo   = (const float*)d_in[10];
  const float* bo   = (const float*)d_in[11];
  float* out = (float*)d_out;

  dim3 grid(B_ROWS / ROWS_PER_BLOCK);
  dim3 block(THREADS);
  hipLaunchKernelGGL(fused_twin_tower_interact, grid, block, 0, stream,
                     x, W1, b1, W2, b2, W3, b3, Wemb, Bemb, Mm, Wo, bo, out);
}
